// SPDANet_32040456028262
// MI455X (gfx1250) — matrix operand order repaired and checked by reading, not compiled, not run on hardware
//
#include <hip/hip_runtime.h>

// ---------------------------------------------------------------------------
// SPDANet for MI455X (gfx1250, wave32, WMMA 16x16x32 f16, f32 accumulate).
// All WMMA operands pre-packed to f16 in memory so GEMM inner loops are pure
// global_load_b128/ds_load -> v_wmma (no v_cvt VALU, no WMMA->VALU WAR nops).
// LSTM x-staging uses CDNA5 async global->LDS DMA (ASYNCcnt) when available.
// ---------------------------------------------------------------------------

typedef __attribute__((ext_vector_type(16))) _Float16 v16h;
typedef __attribute__((ext_vector_type(8)))  float    v8f;
typedef __attribute__((ext_vector_type(4)))  int      v4i;

#define B_    32
#define T_    168
#define D_    512
#define HID_  512
#define GATES_ (4 * HID_)
#define CR_   128
#define H0_   24
#define W0_   7
#define KPAD_ 704      // 168 x-features + 8 pad + 512 h + 16 pad (22 * 32)
#define HOFF_ 176      // offset of h inside the fused K dimension
#define NTIL_X 11      // ceil(168/16)

#if defined(__has_builtin)
#if __has_builtin(__builtin_amdgcn_global_load_async_to_lds_b128) && \
    __has_builtin(__builtin_amdgcn_s_wait_asynccnt)
#define HAS_ASYNC 1
#endif
#endif
#ifndef HAS_ASYNC
#define HAS_ASYNC 0
#endif

__device__ __forceinline__ v8f wmma32(v16h a, v16h b, v8f c) {
  // D = A(16x32 f16) * B(32x16 f16) + C(16x16 f32)
  return __builtin_amdgcn_wmma_f32_16x16x32_f16(
      /*neg_a=*/false, a, /*neg_b=*/false, b,
      /*c_mod=*/(short)0, c, /*reuse_a=*/false, /*reuse_b=*/false);
}

// A fragment from row-major (M x K); lane l: m = l&15, K-half = l>>4; 16 contiguous K per lane.
template <typename T>
__device__ __forceinline__ v16h frag_a_rm(const T* A, int lda, int m0, int k0, int lane) {
  const T* p = A + (size_t)(m0 + (lane & 15)) * lda + k0 + ((lane >> 4) << 4);
  v16h r;
#pragma unroll
  for (int e = 0; e < 8; ++e) r[e] = (_Float16)p[e];
#pragma unroll
  for (int e = 0; e < 8; ++e) r[8 + e] = (_Float16)p[8 + e];
  return r;
}

// B fragment (K x N) stored N-major: W[n, k]; 16 contiguous K per lane.
template <typename T>
__device__ __forceinline__ v16h frag_b_nk(const T* W, int ldk, int n0, int k0, int lane) {
  const T* p = W + (size_t)(n0 + (lane & 15)) * ldk + k0 + ((lane >> 4) << 4);
  v16h r;
#pragma unroll
  for (int e = 0; e < 16; ++e) r[e] = (_Float16)p[e];
  return r;
}

__device__ __forceinline__ float sigf(float x) { return 1.f / (1.f + __expf(-x)); }

// ---------------------------------------------------------------------------
// Prep: f32 -> f16 packing
// ---------------------------------------------------------------------------
__global__ void k_cvt_f16(_Float16* __restrict__ dst, const float* __restrict__ src, int n) {
  int i = blockIdx.x * blockDim.x + threadIdx.x;
  if (i < n) dst[i] = (_Float16)src[i];
}

__global__ void k_pack_wcat(_Float16* __restrict__ dst,
                            const float* __restrict__ wih,   // (2048, 168)
                            const float* __restrict__ whh) { // (2048, 512)
  int i = blockIdx.x * blockDim.x + threadIdx.x;
  if (i >= GATES_ * KPAD_) return;
  int n = i / KPAD_, k = i % KPAD_;
  float v = 0.f;
  if (k < T_) v = wih[n * T_ + k];
  else if (k >= HOFF_ && k < HOFF_ + HID_) v = whh[n * HID_ + (k - HOFF_)];
  dst[i] = (_Float16)v;
}

// ---------------------------------------------------------------------------
// SDK (involution) step, fully f16 operands: one block per batch, 8 waves.
// ---------------------------------------------------------------------------
__global__ void k_sdk_step(const _Float16* __restrict__ Ain, _Float16* __restrict__ Aout,
                           const _Float16* __restrict__ W1h,  // (128,512) f16
                           const float* __restrict__ g1, const float* __restrict__ bb1,
                           const float* __restrict__ W2, const float* __restrict__ b2) {
  extern __shared__ char smem_raw[];
  float* t_s  = (float*)smem_raw;           // 128 x 176
  float* w9_s = t_s + CR_ * 176;            // 9 x 176
  const int b = blockIdx.x, tid = threadIdx.x, lane = tid & 31, w = tid >> 5;
  const _Float16* Ab = Ain + (size_t)b * (T_ * D_);
  const float bnscale = rsqrtf(1.0f + 1e-5f);

  // GEMM1: t = relu(BN(W1 @ x)), x[c,hw] = Ab[hw*512 + c] (128x168, K=512)
  v8f acc[NTIL_X] = {};
  for (int k0 = 0; k0 < D_; k0 += 32) {
    v16h af = frag_a_rm(W1h, D_, w * 16, k0, lane);
#pragma unroll
    for (int n = 0; n < NTIL_X; ++n) {
      int hw = n * 16 + (lane & 15); if (hw >= T_) hw = T_ - 1;  // clamp pad cols
      const _Float16* p = Ab + (size_t)hw * D_ + k0 + ((lane >> 4) << 4);
      v16h bf;
#pragma unroll
      for (int e = 0; e < 16; ++e) bf[e] = p[e];
      acc[n] = wmma32(af, bf, acc[n]);
    }
  }
#pragma unroll
  for (int n = 0; n < NTIL_X; ++n) {
    int hw = n * 16 + (lane & 15);
#pragma unroll
    for (int r = 0; r < 8; ++r) {
      int o = w * 16 + ((lane >> 4) << 3) + r;
      float v = acc[n][r] * (g1[o] * bnscale) + bb1[o];
      t_s[o * 176 + hw] = v > 0.f ? v : 0.f;
    }
  }
  __syncthreads();

  // GEMM2 (tiny): w9[q,hw] = W2 @ t + b2
  for (int idx = tid; idx < 9 * T_; idx += blockDim.x) {
    int q = idx / T_, hw = idx % T_;
    float s = b2[q];
    for (int o = 0; o < CR_; ++o) s += W2[q * CR_ + o] * t_s[o * 176 + hw];
    w9_s[q * 176 + hw] = s;
  }
  __syncthreads();

  // Involution gather + leaky ReLU; Aout flat index c*168+hw (torch .view).
  _Float16* Ao = Aout + (size_t)b * (T_ * D_);
  for (int idx = tid; idx < T_ * D_; idx += blockDim.x) {
    int c = idx & (D_ - 1);
    int hw = idx >> 9;
    int h = hw / W0_, wc = hw % W0_;
    float s = 0.f;
#pragma unroll
    for (int q = 0; q < 9; ++q) {
      int hh = h + q / 3 - 1, ww = wc + q % 3 - 1;
      if (hh >= 0 && hh < H0_ && ww >= 0 && ww < W0_)
        s += w9_s[q * 176 + hw] * (float)Ab[(size_t)(hh * W0_ + ww) * D_ + c];
    }
    Ao[(size_t)c * T_ + hw] = (_Float16)(s > 0.f ? s : 0.01f * s);
  }
}

// seqT[b, d, tau] = sdk_out[b][tau*512 + d]  (f16 -> f16, LSTM time axis = d)
__global__ void k_seqT(_Float16* __restrict__ dst, const _Float16* __restrict__ src) {
  int i = blockIdx.x * blockDim.x + threadIdx.x;
  if (i >= B_ * D_ * T_) return;
  int tau = i % T_;
  int d = (i / T_) % D_;
  int b = i / (T_ * D_);
  dst[i] = src[(size_t)b * (T_ * D_) + (size_t)tau * D_ + d];
}

// ---------------------------------------------------------------------------
// LSTM: one 1024-thread workgroup (32 waves = full WGP) per chain.
// Fused gate GEMM per step: (32 x 704) x (704 -> 2048) in WMMA f16.
// Cell state c lives in VGPRs across all 512 steps; h round-trips via LDS.
// x_{t+1} is prefetched into LDS with async global->LDS DMA (ASYNCcnt),
// overlapped with the gate nonlinearity of step t.
// ---------------------------------------------------------------------------
#if HAS_ASYNC
__device__ __forceinline__ void lstm_stage_async(const _Float16* __restrict__ seqT,
                                                 _Float16* __restrict__ Ash,
                                                 int t, int tid) {
  // 32 rows x 168 halves = 672 chunks of 16 bytes
  if (tid < (B_ * T_) / 8) {
    int bb = tid / (T_ / 8);
    int c  = tid % (T_ / 8);
    const _Float16* g = seqT + (size_t)bb * (D_ * T_) + (size_t)t * T_ + c * 8;
    _Float16* l = Ash + bb * KPAD_ + c * 8;
    __builtin_amdgcn_global_load_async_to_lds_b128(
        (__attribute__((address_space(1))) v4i*)g,
        (__attribute__((address_space(3))) v4i*)l, 0, 0);
  }
}
#endif

__global__ void __launch_bounds__(1024)
k_lstm(const _Float16* __restrict__ seqT,   // (nChain, B, D, T) f16
       const _Float16* __restrict__ Wcat,   // (2048, 704) f16, N-major
       const float* __restrict__ bias,      // (2048)
       float* __restrict__ Hout,            // (nChain, B, HID) f32
       float* __restrict__ Cout,            // (nChain, B, HID) f32
       _Float16* __restrict__ HoutH,        // (nChain, B, HID) f16 (head operand)
       const float* __restrict__ h0,        // nullable (B,HID)
       const float* __restrict__ c0,        // nullable (B,HID)
       const float* __restrict__ ha) {      // nullable, added to h0
  extern __shared__ char smem_raw[];
  _Float16* Ash = (_Float16*)smem_raw;      // (B, KPAD_) fused [x_t | h]
  const int tid = threadIdx.x, lane = tid & 31, w = tid >> 5;
  seqT  += (size_t)blockIdx.x * (B_ * D_ * T_);
  Hout  += (size_t)blockIdx.x * (B_ * HID_);
  Cout  += (size_t)blockIdx.x * (B_ * HID_);
  HoutH += (size_t)blockIdx.x * (B_ * HID_);

  for (int i = tid; i < B_ * KPAD_; i += blockDim.x) Ash[i] = (_Float16)0.f;
  __syncthreads();
  if (h0) {
    for (int i = tid; i < B_ * HID_; i += blockDim.x) {
      int bb = i / HID_, j = i % HID_;
      Ash[bb * KPAD_ + HOFF_ + j] = (_Float16)(h0[i] + (ha ? ha[i] : 0.f));
    }
  }
  const int j0 = w * 16;
  const int j  = j0 + (lane & 15);
  v8f cst[2] = {};
  if (c0) {
#pragma unroll
    for (int m = 0; m < 2; ++m)
#pragma unroll
      for (int r = 0; r < 8; ++r) {
        int bb = m * 16 + ((lane >> 4) << 3) + r;
        cst[m][r] = c0[bb * HID_ + j];
      }
  }
  const float bi  = bias[j];
  const float bfv = bias[HID_ + j];
  const float bg  = bias[2 * HID_ + j];
  const float bo  = bias[3 * HID_ + j];

#if HAS_ASYNC
  lstm_stage_async(seqT, Ash, 0, tid);      // prefetch x_0 (x region only)
#endif

  for (int t = 0; t < D_; ++t) {
#if HAS_ASYNC
    __builtin_amdgcn_s_wait_asynccnt(0);    // x_t landed in LDS
#else
    for (int i = tid; i < B_ * T_; i += blockDim.x) {
      int bb = i / T_, tau = i % T_;
      Ash[bb * KPAD_ + tau] = seqT[(size_t)bb * (D_ * T_) + (size_t)t * T_ + tau];
    }
#endif
    __syncthreads();                        // x_t + h_t visible to all waves

    v8f acc[4][2] = {};
    for (int k0 = 0; k0 < KPAD_; k0 += 32) {
      v16h a0 = frag_a_rm(Ash, KPAD_, 0,  k0, lane);
      v16h a1 = frag_a_rm(Ash, KPAD_, 16, k0, lane);
#pragma unroll
      for (int g = 0; g < 4; ++g) {
        v16h bf = frag_b_nk(Wcat, KPAD_, g * HID_ + j0, k0, lane);
        acc[g][0] = wmma32(a0, bf, acc[g][0]);
        acc[g][1] = wmma32(a1, bf, acc[g][1]);
      }
    }
    __syncthreads();                        // all reads of Ash complete

#if HAS_ASYNC
    if (t + 1 < D_) lstm_stage_async(seqT, Ash, t + 1, tid);  // overlap w/ gates
#endif

#pragma unroll
    for (int m = 0; m < 2; ++m)
#pragma unroll
      for (int r = 0; r < 8; ++r) {
        int bb = m * 16 + ((lane >> 4) << 3) + r;
        float iv = sigf(acc[0][m][r] + bi);
        float fv = sigf(acc[1][m][r] + bfv);
        float gv = tanhf(acc[2][m][r] + bg);
        float ov = sigf(acc[3][m][r] + bo);
        float cn = fv * cst[m][r] + iv * gv;
        cst[m][r] = cn;
        float hn = ov * tanhf(cn);
        Ash[bb * KPAD_ + HOFF_ + j] = (_Float16)hn;
        if (t == D_ - 1) {
          Hout[bb * HID_ + j]  = hn;
          Cout[bb * HID_ + j]  = cn;
          HoutH[bb * HID_ + j] = (_Float16)hn;
        }
      }
  }
}

// ---------------------------------------------------------------------------
// Head: mix[:,0:512] = Hs[s] @ Wl[s]^T + bl[s] + horizon terms  (f16 operands)
// ---------------------------------------------------------------------------
__global__ void k_head(_Float16* __restrict__ mixh, const _Float16* __restrict__ HallH,
                       const _Float16* __restrict__ Wlh, const float* __restrict__ bl,
                       const float* __restrict__ X, const float* __restrict__ hw_w,
                       const float* __restrict__ hw_b) {
  const int s = blockIdx.x, tid = threadIdx.x, lane = tid & 31, w = tid >> 5;
  const _Float16* Hsrc = HallH + (size_t)s * B_ * HID_;   // [sh0, th, sh2]
  const _Float16* Wls  = Wlh + (size_t)s * D_ * HID_;     // (d, h) N-major
  v8f acc[2][4] = {};
  for (int k0 = 0; k0 < HID_; k0 += 32) {
    v16h a0 = frag_a_rm(Hsrc, HID_, 0,  k0, lane);
    v16h a1 = frag_a_rm(Hsrc, HID_, 16, k0, lane);
#pragma unroll
    for (int q = 0; q < 4; ++q) {
      v16h bf = frag_b_nk(Wls, HID_, (w * 4 + q) * 16, k0, lane);
      acc[0][q] = wmma32(a0, bf, acc[0][q]);
      acc[1][q] = wmma32(a1, bf, acc[1][q]);
    }
  }
  const int Lw = 2 * (s + 1);
#pragma unroll
  for (int mt = 0; mt < 2; ++mt)
#pragma unroll
    for (int q = 0; q < 4; ++q) {
      int d = (w * 4 + q) * 16 + (lane & 15);
#pragma unroll
      for (int r = 0; r < 8; ++r) {
        int bb = mt * 16 + ((lane >> 4) << 3) + r;
        float v = acc[mt][q][r] + bl[s * D_ + d] + hw_b[s];
        for (int l = 0; l < Lw; ++l)
          v += X[(size_t)bb * (T_ * D_) + (size_t)(T_ - Lw + l) * D_ + d] * hw_w[s * 6 + l];
        mixh[(size_t)(bb * 3 + s) * 1024 + d] = (_Float16)v;
      }
    }
}

__global__ void k_fill_nppt(_Float16* __restrict__ mixh, const float* __restrict__ nppt) {
  int i = blockIdx.x * blockDim.x + threadIdx.x;
  if (i >= B_ * 3 * D_) return;
  int d = i % D_;
  int s = (i / D_) % 3;
  int b = i / (3 * D_);
  mixh[(size_t)(b * 3 + s) * 1024 + D_ + d] = (_Float16)nppt[i];
}

// FFN1: (96x1024 f16) @ Wf1h^T -> relu -> f16 (96x1024)
__global__ void k_ffn1(_Float16* __restrict__ H1, const _Float16* __restrict__ mixh,
                       const _Float16* __restrict__ Wf1h, const float* __restrict__ bf1) {
  const int tid = threadIdx.x, lane = tid & 31, w = tid >> 5;
  const int mt = blockIdx.x % 6;
  const int nh = blockIdx.x / 6;
  v8f acc[4] = {};
  for (int k0 = 0; k0 < 1024; k0 += 32) {
    v16h a = frag_a_rm(mixh, 1024, mt * 16, k0, lane);
#pragma unroll
    for (int q = 0; q < 4; ++q) {
      v16h bf = frag_b_nk(Wf1h, 1024, (nh * 32 + w * 4 + q) * 16, k0, lane);
      acc[q] = wmma32(a, bf, acc[q]);
    }
  }
#pragma unroll
  for (int q = 0; q < 4; ++q) {
    int n = (nh * 32 + w * 4 + q) * 16 + (lane & 15);
#pragma unroll
    for (int r = 0; r < 8; ++r) {
      int row = mt * 16 + ((lane >> 4) << 3) + r;
      float v = acc[q][r] + bf1[n];
      H1[(size_t)row * 1024 + n] = (_Float16)(v > 0.f ? v : 0.f);
    }
  }
}

// FFN2: (96x1024 f16) @ Wf2h^T + bf2 -> d_out (B,S,D) f32
__global__ void k_ffn2(float* __restrict__ out, const _Float16* __restrict__ H1,
                       const _Float16* __restrict__ Wf2h, const float* __restrict__ bf2) {
  const int tid = threadIdx.x, lane = tid & 31, w = tid >> 5;
  const int mt = blockIdx.x;
  v8f acc[4] = {};
  for (int k0 = 0; k0 < 1024; k0 += 32) {
    v16h a = frag_a_rm(H1, 1024, mt * 16, k0, lane);
#pragma unroll
    for (int q = 0; q < 4; ++q) {
      v16h bf = frag_b_nk(Wf2h, 1024, (w * 4 + q) * 16, k0, lane);
      acc[q] = wmma32(a, bf, acc[q]);
    }
  }
#pragma unroll
  for (int q = 0; q < 4; ++q) {
    int n = (w * 4 + q) * 16 + (lane & 15);
#pragma unroll
    for (int r = 0; r < 8; ++r) {
      int row = mt * 16 + ((lane >> 4) << 3) + r;
      out[(size_t)row * D_ + n] = acc[q][r] + bf2[n];
    }
  }
}

// ---------------------------------------------------------------------------
extern "C" void kernel_launch(void* const* d_in, const int* in_sizes, int n_in,
                              void* d_out, int out_size, void* d_ws, size_t ws_size,
                              hipStream_t stream) {
  (void)in_sizes; (void)n_in; (void)out_size; (void)ws_size;
  const float* X     = (const float*)d_in[0];
  const float* NPPT  = (const float*)d_in[1];
  const float* ha    = (const float*)d_in[2];
  const float* W1    = (const float*)d_in[3];
  const float* g1    = (const float*)d_in[4];
  const float* bb1   = (const float*)d_in[5];
  const float* W2    = (const float*)d_in[6];
  const float* b2    = (const float*)d_in[7];
  const float* Wih_s = (const float*)d_in[8];
  const float* Whh_s = (const float*)d_in[9];
  const float* b_s   = (const float*)d_in[10];
  const float* Wih_t = (const float*)d_in[11];
  const float* Whh_t = (const float*)d_in[12];
  const float* b_t   = (const float*)d_in[13];
  const float* Wl    = (const float*)d_in[14];
  const float* bl    = (const float*)d_in[15];
  const float* hw_w  = (const float*)d_in[16];
  const float* hw_b  = (const float*)d_in[17];
  const float* Wf1   = (const float*)d_in[18];
  const float* bf1   = (const float*)d_in[19];
  const float* Wf2   = (const float*)d_in[20];
  const float* bf2   = (const float*)d_in[21];

  size_t off = 0;
  auto carve = [&](size_t bytes) -> void* {
    void* q = (char*)d_ws + off;
    off += (bytes + 255) & ~(size_t)255;
    return q;
  };
  const size_t slab = (size_t)B_ * T_ * D_;                       // 2,752,512 elems
  _Float16* Xh     = (_Float16*)carve(slab * 2);                  // f16 copy of X
  _Float16* sdkOut = (_Float16*)carve(3 * slab * 2);              // sdk_outs[0..2] f16
  _Float16* seqT   = (_Float16*)carve(3 * slab * 2);              // (k,B,D,T) f16
  _Float16* WcatS  = (_Float16*)carve((size_t)GATES_ * KPAD_ * 2);
  _Float16* WcatT  = (_Float16*)carve((size_t)GATES_ * KPAD_ * 2);
  _Float16* W1h    = (_Float16*)carve((size_t)CR_ * D_ * 2);
  _Float16* Wlh    = (_Float16*)carve((size_t)3 * D_ * HID_ * 2);
  _Float16* Wf1h   = (_Float16*)carve((size_t)1024 * 1024 * 2);
  _Float16* Wf2h   = (_Float16*)carve((size_t)D_ * 1024 * 2);
  float*    Hsh    = (float*)carve((size_t)3 * B_ * HID_ * 4);    // shared h (f32)
  float*    Csh    = (float*)carve((size_t)3 * B_ * HID_ * 4);    // shared c (f32)
  float*    Hdmp   = (float*)carve((size_t)B_ * HID_ * 4);        // target h (unused)
  float*    Cdmp   = (float*)carve((size_t)B_ * HID_ * 4);        // target c (unused)
  _Float16* HallH  = (_Float16*)carve((size_t)3 * B_ * HID_ * 2); // [sh0, th, sh2] f16
  _Float16* mixh   = (_Float16*)carve((size_t)96 * 1024 * 2);
  _Float16* H1     = (_Float16*)carve((size_t)96 * 1024 * 2);

  // --- weight/activation prep (f16 packing) ---
  auto cvt = [&](_Float16* dst, const float* src, int n) {
    k_cvt_f16<<<(n + 255) / 256, 256, 0, stream>>>(dst, src, n);
  };
  cvt(Xh, X, (int)slab);
  cvt(W1h, W1, CR_ * D_);
  cvt(Wlh, Wl, 3 * D_ * HID_);
  cvt(Wf1h, Wf1, 1024 * 1024);
  cvt(Wf2h, Wf2, D_ * 1024);
  int nW = GATES_ * KPAD_;
  k_pack_wcat<<<(nW + 255) / 256, 256, 0, stream>>>(WcatS, Wih_s, Whh_s);
  k_pack_wcat<<<(nW + 255) / 256, 256, 0, stream>>>(WcatT, Wih_t, Whh_t);

  // --- SDK chain (3 involution steps, f16 in / f16 out) ---
  size_t sdkLds = (size_t)(CR_ * 176 + 9 * 176) * sizeof(float);
  k_sdk_step<<<B_, 256, sdkLds, stream>>>(Xh, sdkOut, W1h, g1, bb1, W2, b2);
  k_sdk_step<<<B_, 256, sdkLds, stream>>>(sdkOut, sdkOut + slab, W1h, g1, bb1, W2, b2);
  k_sdk_step<<<B_, 256, sdkLds, stream>>>(sdkOut + slab, sdkOut + 2 * slab, W1h, g1, bb1, W2, b2);

  // --- transpose to (B, D, T) f16 for LSTM staging ---
  int nT = B_ * D_ * T_;
  for (int k = 0; k < 3; ++k)
    k_seqT<<<(nT + 255) / 256, 256, 0, stream>>>(seqT + (size_t)k * slab,
                                                 sdkOut + (size_t)k * slab);

  // --- LSTMs: 3 shared chains (3 workgroups), then target chain ---
  size_t lstmLds = (size_t)B_ * KPAD_ * sizeof(_Float16);
  k_lstm<<<3, 1024, lstmLds, stream>>>(seqT, WcatS, b_s, Hsh, Csh, HallH,
                                       nullptr, nullptr, nullptr);
  k_lstm<<<1, 1024, lstmLds, stream>>>(seqT + slab, WcatT, b_t, Hdmp, Cdmp,
                                       HallH + (size_t)B_ * HID_,   // th -> slot 1
                                       Hsh + (size_t)B_ * HID_,
                                       Csh + (size_t)B_ * HID_, ha);

  // --- head + FFN (pure f16 WMMA operands) ---
  k_head<<<3, 256, 0, stream>>>(mixh, HallH, Wlh, bl, X, hw_w, hw_b);
  k_fill_nppt<<<(B_ * 3 * D_ + 255) / 256, 256, 0, stream>>>(mixh, NPPT);
  k_ffn1<<<12, 256, 0, stream>>>(H1, mixh, Wf1h, bf1);
  k_ffn2<<<6, 256, 0, stream>>>((float*)d_out, H1, Wf2h, bf2);
}
